// PFNet6_16767552323984
// MI455X (gfx1250) — compile-verified, matrix operand order checked by reading, not run on hardware
//
#include <hip/hip_runtime.h>

// ---------------------------------------------------------------------------
// PFNet6 for MI455X (gfx1250, wave32). Dense layers -> v_wmma_f32_16x16x32_f16
// with persistent waves + register-resident weight fragments. Graph scatter ->
// f32 global atomics (all tables L2-resident on the 192MB L2).
// ---------------------------------------------------------------------------

typedef __attribute__((ext_vector_type(16))) _Float16 v16h;
typedef __attribute__((ext_vector_type(8)))  float    v8f;

// branchless SELU: selu(v) = s*(max(v,0) + a*(exp(min(v,0))-1))
__device__ __forceinline__ float selu_f(float v) {
  const float scale = 1.0507009873554804934193349852946f;
  const float alpha = 1.6732632423543772848170429916717f;
  return scale * (fmaxf(v, 0.0f) + alpha * (__expf(fminf(v, 0.0f)) - 1.0f));
}

// 16-bit A-matrix 16x32 striping (ISA 7.12.2): lane = M row (mod 16),
// half = lane>>4 selects K sub-bank. Element j of the v16h fragment maps to:
__device__ __forceinline__ int frag_k(int j, int half) {
  return (j < 8) ? (j + 8 * half) : (j + 8 + 8 * half);
}

template <int STRIDE>
__device__ __forceinline__ v16h load_a_frag(const float* t, int k0, int lane) {
  const int m = lane & 15, half = (lane >> 4) & 1;
  v16h a;
#pragma unroll
  for (int j = 0; j < 16; ++j)
    a[j] = (_Float16)t[m * STRIDE + k0 + frag_k(j, half)];
  return a;
}

// B (KxN) column-striped: lane = N column (mod 16), same K distribution.
// W stored row-major (din, dout). Branchless K/N zero-padding (clamp+select,
// no exec-mask divergence around the loads).
template <int DIN, int DOUT>
__device__ __forceinline__ v16h load_b_frag(const float* W, int k0, int n0, int lane) {
  const int n = n0 + (lane & 15), half = (lane >> 4) & 1;
  const int nc = (n < DOUT) ? n : 0;
  v16h b;
#pragma unroll
  for (int j = 0; j < 16; ++j) {
    const int k = k0 + frag_k(j, half);
    const int kc = (k < DIN) ? k : 0;
    float w = W[kc * DOUT + nc];
    if (k >= DIN || n >= DOUT) w = 0.0f;
    b[j] = (_Float16)w;
  }
  return b;
}

// register-resident layer weights: NT*KT B-fragments + per-lane bias values
template <int DIN, int DOUT>
struct LayerW {
  static constexpr int NT = (DOUT + 15) / 16;
  static constexpr int KT = (DIN + 31) / 32;
  v16h  wb[NT][KT];
  float bv[NT];
};

template <int DIN, int DOUT>
__device__ __forceinline__ LayerW<DIN, DOUT> load_layer(const float* W, const float* bias,
                                                        int lane) {
  LayerW<DIN, DOUT> L;
  const int nl = lane & 15;
#pragma unroll
  for (int nt = 0; nt < L.NT; ++nt) {
    const int n = nt * 16 + nl;
    float bvv = bias[(n < DOUT) ? n : 0];
    L.bv[nt] = (n < DOUT) ? bvv : 0.0f;
#pragma unroll
    for (int kt = 0; kt < L.KT; ++kt)
      L.wb[nt][kt] = load_b_frag<DIN, DOUT>(W, kt * 32, nt * 16, lane);
  }
  return L;
}

// One dense layer on a 16-row LDS tile: tout = act(tin @ W + b).
// tin stride = STRIDE (K padded), tout stride = 32. In-place safe: all A
// fragments are read before any write (DS ops are in-order within a wave).
template <int DIN, int DOUT, int STRIDE, bool ACT>
__device__ __forceinline__ void apply_layer(const float* tin, const LayerW<DIN, DOUT>& L,
                                            float* tout, int lane) {
  constexpr int NT = (DOUT + 15) / 16;
  constexpr int KT = (DIN + 31) / 32;
  const int n = lane & 15, mb = (lane >> 4) << 3;
  v16h a[KT];
#pragma unroll
  for (int kt = 0; kt < KT; ++kt) a[kt] = load_a_frag<STRIDE>(tin, kt << 5, lane);
  v8f acc[NT];
#pragma unroll
  for (int nt = 0; nt < NT; ++nt) {
    v8f c;
#pragma unroll
    for (int v = 0; v < 8; ++v) c[v] = L.bv[nt];
#pragma unroll
    for (int kt = 0; kt < KT; ++kt)
      c = __builtin_amdgcn_wmma_f32_16x16x32_f16(false, a[kt], false, L.wb[nt][kt],
                                                 (short)0, c, false, false);
    acc[nt] = c;
  }
#pragma unroll
  for (int nt = 0; nt < NT; ++nt) {
#pragma unroll
    for (int v = 0; v < 8; ++v) {
      float r = acc[nt][v];
      if (ACT) r = selu_f(r);
      tout[(mb + v) * 32 + (nt << 4) + n] = r;   // C/D layout: M=v+8*(lane>>4), N=lane&15
    }
  }
}

// --------------------------- kernels ---------------------------------------

struct MLP4P { const float *W0,*b0,*W1,*b1,*W2,*b2,*W3,*b3; };
struct EdgeP { const float *W0,*b0,*W1,*b1,*W2,*b2; };
struct HeadP {
  const float *a0W,*a0b,*a1W,*a1b,*a2W,*a2b,*a3W,*a3b,*a4W,*a4b;   // nn1
  const float *c0W,*c0b,*c1W,*c1b,*c2W,*c2b,*c3W,*c3b,*c4W,*c4b;   // nn2
};

// input MLP: x (N,15) -> h (N,32), 4 SELU layers. Persistent waves: weight
// fragments live in VGPRs across the grid-stride tile loop. launch_bounds
// (·,1) relaxes the occupancy target so no fragment spills to scratch.
__global__ void __launch_bounds__(256, 1)
k_node_mlp4(const float* __restrict__ x, MLP4P p,
            float* __restrict__ hout, int nNodes, int nTiles) {
  __shared__ float tiles[8][16 * 32];
  const int lane = threadIdx.x & 31, wave = threadIdx.x >> 5;
  float* t = tiles[wave];
  const LayerW<15, 32> L0 = load_layer<15, 32>(p.W0, p.b0, lane);
  const LayerW<32, 32> L1 = load_layer<32, 32>(p.W1, p.b1, lane);
  const LayerW<32, 32> L2 = load_layer<32, 32>(p.W2, p.b2, lane);
  const LayerW<32, 32> L3 = load_layer<32, 32>(p.W3, p.b3, lane);
  for (int tile = blockIdx.x * 8 + wave; tile < nTiles; tile += gridDim.x * 8) {
    const int base = tile * 16;
    if (lane < 16) {
      const int node = base + lane;
      const bool ok = node < nNodes;
#pragma unroll
      for (int k = 0; k < 15; ++k) t[lane * 32 + k] = ok ? x[node * 15 + k] : 0.0f;
#pragma unroll
      for (int k = 15; k < 32; ++k) t[lane * 32 + k] = 0.0f;
    }
    apply_layer<15, 32, 32, true>(t, L0, t, lane);
    apply_layer<32, 32, 32, true>(t, L1, t, lane);
    apply_layer<32, 32, 32, true>(t, L2, t, lane);
    apply_layer<32, 32, 32, true>(t, L3, t, lane);
    if (lane < 16) {
      const int node = base + lane;
      if (node < nNodes)
#pragma unroll
        for (int k = 0; k < 32; ++k) hout[node * 32 + k] = t[lane * 32 + k];
    }
  }
}

// single 32x32 layer + SELU (conv linear), persistent waves
__global__ void __launch_bounds__(256, 1)
k_node_linear(const float* __restrict__ hin, const float* __restrict__ W,
              const float* __restrict__ b, float* __restrict__ hout,
              int nNodes, int nTiles) {
  __shared__ float tiles[8][16 * 32];
  const int lane = threadIdx.x & 31, wave = threadIdx.x >> 5;
  float* t = tiles[wave];
  const LayerW<32, 32> L = load_layer<32, 32>(W, b, lane);
  for (int tile = blockIdx.x * 8 + wave; tile < nTiles; tile += gridDim.x * 8) {
    const int base = tile * 16;
    if (lane < 16) {
      const int node = base + lane;
      const bool ok = node < nNodes;
#pragma unroll
      for (int k = 0; k < 32; ++k) t[lane * 32 + k] = ok ? hin[node * 32 + k] : 0.0f;
    }
    apply_layer<32, 32, 32, true>(t, L, t, lane);
    if (lane < 16) {
      const int node = base + lane;
      if (node < nNodes)
#pragma unroll
        for (int k = 0; k < 32; ++k) hout[node * 32 + k] = t[lane * 32 + k];
    }
  }
}

// edge MLP: [h[row,:2], h[col,:2], ew] (5 feats, K-padded to 32) -> sigmoid.
// Persistent waves; all three layers (incl. 32->1) run on the WMMA path.
__global__ void __launch_bounds__(256, 1)
k_edge_mlp(const float* __restrict__ h, const int* __restrict__ row,
           const int* __restrict__ col, const float* __restrict__ ea,
           EdgeP p, float* __restrict__ ew2, int nEdges, int nTiles) {
  __shared__ float tiles[8][16 * 32];
  const int lane = threadIdx.x & 31, wave = threadIdx.x >> 5;
  float* t = tiles[wave];
  const LayerW<5, 32>  L0 = load_layer<5, 32>(p.W0, p.b0, lane);
  const LayerW<32, 32> L1 = load_layer<32, 32>(p.W1, p.b1, lane);
  const LayerW<32, 1>  L2 = load_layer<32, 1>(p.W2, p.b2, lane);
  for (int tile = blockIdx.x * 8 + wave; tile < nTiles; tile += gridDim.x * 8) {
    const int base = tile * 16;
    if (lane < 16) {
      const int e = base + lane;
      const bool ok = e < nEdges;
      const int r = ok ? row[e] : 0, c = ok ? col[e] : 0;
      t[lane * 32 + 0] = ok ? h[r * 32 + 0] : 0.0f;
      t[lane * 32 + 1] = ok ? h[r * 32 + 1] : 0.0f;
      t[lane * 32 + 2] = ok ? h[c * 32 + 0] : 0.0f;
      t[lane * 32 + 3] = ok ? h[c * 32 + 1] : 0.0f;
      t[lane * 32 + 4] = ok ? ea[e] : 0.0f;
#pragma unroll
      for (int k = 5; k < 32; ++k) t[lane * 32 + k] = 0.0f;
    }
    apply_layer<5, 32, 32, true >(t, L0, t, lane);
    apply_layer<32, 32, 32, true >(t, L1, t, lane);
    apply_layer<32,  1, 32, false>(t, L2, t, lane);   // logit lands in col 0
    if (lane < 16) {
      const int e = base + lane;
      if (e < nEdges) {
        const float s = t[lane * 32];
        ew2[e] = 1.0f / (1.0f + __expf(-s));
      }
    }
  }
}

__global__ void k_fill(float* __restrict__ p, float v, int n) {
  const int i = blockIdx.x * blockDim.x + threadIdx.x;
  if (i < n) p[i] = v;
}

__global__ void k_deg(const int* __restrict__ col, const float* __restrict__ ew2,
                      int useW2, float* __restrict__ deg, int nEdges) {
  const int e = blockIdx.x * blockDim.x + threadIdx.x;
  if (e >= nEdges) return;
  const float w = useW2 ? ((ew2[e] > 0.5f) ? 1.0f : 0.0f) : 1.0f;
  if (w != 0.0f) atomicAdd(&deg[col[e]], w);
}

__global__ void k_rsqrt(const float* __restrict__ deg, float* __restrict__ dinv, int n) {
  const int i = blockIdx.x * blockDim.x + threadIdx.x;
  if (i < n) dinv[i] = rsqrtf(deg[i]);
}

// out = h * dinv^2 (self term of SGConv iteration)
__global__ void k_self(const float* __restrict__ h, const float* __restrict__ dinv,
                       float* __restrict__ out, int nElems) {
  const int i = blockIdx.x * blockDim.x + threadIdx.x;
  if (i >= nElems) return;
  const float d = dinv[i >> 5];
  out[i] = h[i] * d * d;
}

// out[col] += h[row] * dinv[row]*w*dinv[col]  (f32 atomic scatter, L2-resident)
__global__ void k_prop(const float* __restrict__ h, const int* __restrict__ row,
                       const int* __restrict__ col, const float* __restrict__ dinv,
                       const float* __restrict__ ew2, int useW2,
                       float* __restrict__ out, int nEdges) {
  const int e = blockIdx.x * blockDim.x + threadIdx.x;
  if (e >= nEdges) return;
  const float w = useW2 ? ((ew2[e] > 0.5f) ? 1.0f : 0.0f) : 1.0f;
  if (w == 0.0f) return;
  const int r = row[e], c = col[e];
  const float norm = dinv[r] * w * dinv[c];
  const float4* hs = (const float4*)(h + (size_t)r * 32);
  float* od = out + (size_t)c * 32;
#pragma unroll
  for (int q = 0; q < 8; ++q) {
    const float4 v = hs[q];
    atomicAdd(od + q * 4 + 0, v.x * norm);
    atomicAdd(od + q * 4 + 1, v.y * norm);
    atomicAdd(od + q * 4 + 2, v.z * norm);
    atomicAdd(od + q * 4 + 3, v.w * norm);
  }
}

// fused nn1 + nn2 heads; 128-thread blocks (4 waves) -> 32KB static LDS.
// Only 6250 tiles total, so weights are (re)loaded per tile; launch_bounds
// (·,1) still lets live ranges stay in registers without spilling.
__global__ void __launch_bounds__(128, 1)
k_heads(const float* __restrict__ h, const float* __restrict__ x,
        HeadP p, float* __restrict__ outIds, float* __restrict__ outP4,
        int nNodes) {
  __shared__ float tH[4][16 * 32];
  __shared__ float tW[4][16 * 32];
  __shared__ float t64[4][16 * 64];
  const int lane = threadIdx.x & 31, wave = threadIdx.x >> 5;
  const int base = (blockIdx.x * 4 + wave) * 16;
  if (base >= nNodes) return;
  float* th = tH[wave];
  float* tw = tW[wave];
  float* t6 = t64[wave];
  if (lane < 16) {
    const int node = base + lane;
    const bool ok = node < nNodes;
#pragma unroll
    for (int k = 0; k < 32; ++k) th[lane * 32 + k] = ok ? h[node * 32 + k] : 0.0f;
  }
  // nn1: 4x SELU + linear 32->6
  apply_layer<32, 32, 32, true >(th, load_layer<32, 32>(p.a0W, p.a0b, lane), tw, lane);
  apply_layer<32, 32, 32, true >(tw, load_layer<32, 32>(p.a1W, p.a1b, lane), tw, lane);
  apply_layer<32, 32, 32, true >(tw, load_layer<32, 32>(p.a2W, p.a2b, lane), tw, lane);
  apply_layer<32, 32, 32, true >(tw, load_layer<32, 32>(p.a3W, p.a3b, lane), tw, lane);
  apply_layer<32,  6, 32, false>(tw, load_layer<32, 6>(p.a4W, p.a4b, lane), tw, lane);
  if (lane < 16) {
    const int node = base + lane;
    const bool ok = node < nNodes;
#pragma unroll
    for (int k = 0; k < 6; ++k) {             // nn2 input: [cand_ids(6), h(32), 0-pad]
      const float v = tw[lane * 32 + k];
      t6[lane * 64 + k] = v;
      if (ok) outIds[(size_t)node * 6 + k] = v;
    }
#pragma unroll
    for (int k = 0; k < 32; ++k) t6[lane * 64 + 6 + k] = th[lane * 32 + k];
#pragma unroll
    for (int k = 38; k < 64; ++k) t6[lane * 64 + k] = 0.0f;
  }
  // nn2: K=38 padded to 64 (two accumulating WMMAs), 3x SELU, linear 32->4
  apply_layer<38, 32, 64, true >(t6, load_layer<38, 32>(p.c0W, p.c0b, lane), tw, lane);
  apply_layer<32, 32, 32, true >(tw, load_layer<32, 32>(p.c1W, p.c1b, lane), tw, lane);
  apply_layer<32, 32, 32, true >(tw, load_layer<32, 32>(p.c2W, p.c2b, lane), tw, lane);
  apply_layer<32, 32, 32, true >(tw, load_layer<32, 32>(p.c3W, p.c3b, lane), tw, lane);
  apply_layer<32,  4, 32, false>(tw, load_layer<32, 4>(p.c4W, p.c4b, lane), tw, lane);
  if (lane < 16) {
    const int node = base + lane;
    if (node < nNodes)
#pragma unroll
      for (int k = 0; k < 4; ++k)
        outP4[(size_t)node * 4 + k] = x[node * 15 + 3 + k] + tw[lane * 32 + k];
  }
}

// --------------------------- launcher --------------------------------------

extern "C" void kernel_launch(void* const* d_in, const int* in_sizes, int n_in,
                              void* d_out, int out_size, void* d_ws, size_t ws_size,
                              hipStream_t stream) {
  (void)n_in; (void)out_size; (void)ws_size;
  const float* x  = (const float*)d_in[0];
  const int*   ei = (const int*)d_in[1];
  const float* ea = (const float*)d_in[2];
  const int N = in_sizes[0] / 15;
  const int E = in_sizes[2];
  const int* row = ei;
  const int* col = ei + E;

  #define FP(i) ((const float*)d_in[(i)])
  MLP4P inp { FP(3), FP(4), FP(5), FP(6), FP(7), FP(8), FP(9), FP(10) };
  const float* conv1W = FP(11); const float* conv1b = FP(12);
  EdgeP ep { FP(13), FP(14), FP(15), FP(16), FP(17), FP(18) };
  const float* conv2W = FP(19); const float* conv2b = FP(20);
  HeadP hp { FP(21), FP(22), FP(23), FP(24), FP(25), FP(26), FP(27), FP(28), FP(29), FP(30),
             FP(31), FP(32), FP(33), FP(34), FP(35), FP(36), FP(37), FP(38), FP(39), FP(40) };
  #undef FP

  // workspace: 3 node-feature buffers + deg + dinv
  float* hA   = (float*)d_ws;
  float* hB   = hA + (size_t)N * 32;
  float* hC   = hB + (size_t)N * 32;
  float* deg  = hC + (size_t)N * 32;
  float* dinv = deg + N;

  // d_out = [ew2(E) | cand_ids(N*6) | cand_p4(N*4)]
  float* ew2    = (float*)d_out;
  float* outIds = ew2 + (size_t)E;
  float* outP4  = outIds + (size_t)N * 6;

  const int nodeTiles = (N + 15) / 16;
  const int edgeTiles = (E + 15) / 16;
  int gNodeTile = (nodeTiles + 7) / 8; if (gNodeTile > 1024) gNodeTile = 1024;
  int gEdgeTile = (edgeTiles + 7) / 8; if (gEdgeTile > 2048) gEdgeTile = 2048;
  const int gHead = (nodeTiles + 3) / 4;
  const int gN  = (N + 255) / 256;
  const int gE  = (E + 255) / 256;
  const int gNF = (N * 32 + 255) / 256;

  // 1. input MLP
  k_node_mlp4<<<gNodeTile, 256, 0, stream>>>(x, inp, hA, N, nodeTiles);

  // 2. conv1 (w = 1)
  k_fill <<<gN, 256, 0, stream>>>(deg, 1.0f, N);
  k_deg  <<<gE, 256, 0, stream>>>(col, ew2, 0, deg, E);
  k_rsqrt<<<gN, 256, 0, stream>>>(deg, dinv, N);
  k_self <<<gNF, 256, 0, stream>>>(hA, dinv, hB, N * 32);
  k_prop <<<gE, 256, 0, stream>>>(hA, row, col, dinv, ew2, 0, hB, E);
  k_self <<<gNF, 256, 0, stream>>>(hB, dinv, hC, N * 32);
  k_prop <<<gE, 256, 0, stream>>>(hB, row, col, dinv, ew2, 0, hC, E);
  k_node_linear<<<gNodeTile, 256, 0, stream>>>(hC, conv1W, conv1b, hA, N, nodeTiles);

  // 3. edge MLP -> ew2 (output; also re-used as conv2 edge weight mask)
  k_edge_mlp<<<gEdgeTile, 256, 0, stream>>>(hA, row, col, ea, ep, ew2, E, edgeTiles);

  // 4. conv2 (w = ew2 > 0.5)
  k_fill <<<gN, 256, 0, stream>>>(deg, 1.0f, N);
  k_deg  <<<gE, 256, 0, stream>>>(col, ew2, 1, deg, E);
  k_rsqrt<<<gN, 256, 0, stream>>>(deg, dinv, N);
  k_self <<<gNF, 256, 0, stream>>>(hA, dinv, hB, N * 32);
  k_prop <<<gE, 256, 0, stream>>>(hA, row, col, dinv, ew2, 1, hB, E);
  k_self <<<gNF, 256, 0, stream>>>(hB, dinv, hC, N * 32);
  k_prop <<<gE, 256, 0, stream>>>(hB, row, col, dinv, ew2, 1, hC, E);
  k_node_linear<<<gNodeTile, 256, 0, stream>>>(hC, conv2W, conv2b, hB, N, nodeTiles);

  // 5. fused nn1/nn2 heads
  k_heads<<<gHead, 128, 0, stream>>>(hB, x, hp, outIds, outP4, N);
}
